// FCAtenttion_86234353369623
// MI455X (gfx1250) — matrix operand order repaired and checked by reading, not compiled, not run on hardware
//
#include <hip/hip_runtime.h>
#include <hip/hip_bf16.h>

// ---------------------------------------------------------------------------
// Problem constants (from reference): B=16, S=2048, D_IN=512, D_H=1024, D_OUT=10
// ---------------------------------------------------------------------------
#define CB    16
#define CS    2048
#define CDIN  512
#define CDH   1024
#define CDOUT 10

typedef __attribute__((ext_vector_type(16))) __bf16 v16bf;
typedef __attribute__((ext_vector_type(8)))  float  v8f;

union BF16Frag { v16bf v; unsigned int u[8]; };

__device__ __forceinline__ unsigned short f2bf(float x) {
    unsigned u = __builtin_bit_cast(unsigned, x);
    unsigned r = u + 0x7FFFu + ((u >> 16) & 1u);   // round-to-nearest-even
    return (unsigned short)(r >> 16);
}

// ---------------------------------------------------------------------------
// Tensor Data Mover support (probe-verified builtin; arity differs between
// toolchains -- the TDM header ships only with the 6-arg toolchain).
// ---------------------------------------------------------------------------
#if defined(__HIP_DEVICE_COMPILE__) && defined(__gfx1250__) && \
    __has_builtin(__builtin_amdgcn_tensor_load_to_lds) && \
    __has_builtin(__builtin_amdgcn_s_wait_tensorcnt)
#define HAVE_TDM 1
#else
#define HAVE_TDM 0
#endif

#if HAVE_TDM && __has_include(<hip/amd_detail/amd_gfx1250_TDM.h>)
#define TDM_ARITY6 1
#else
#define TDM_ARITY6 0
#endif

#if HAVE_TDM
typedef __attribute__((ext_vector_type(4))) unsigned int tdm_u32x4;
typedef __attribute__((ext_vector_type(8))) int          tdm_i32x8;
typedef __attribute__((ext_vector_type(4))) int          tdm_i32x4;

// 2D tile load: tile_k x tile_rows of 2-byte elements, row stride in elements.
// D# bit layout per CDNA5 ISA ch.8 (group0: count/lds_addr/global_addr/type,
// group1: data_size, tensor dims, tile dims, dim0 stride).
__device__ __forceinline__ void tdm_load_tile_2d(unsigned lds_byte_off,
                                                 const void* gptr,
                                                 unsigned tile_k, unsigned tile_rows,
                                                 unsigned tensor_k, unsigned tensor_rows,
                                                 unsigned long long row_stride_elems) {
    unsigned long long ga = (unsigned long long)gptr;
    tdm_u32x4 g0;
    g0[0] = 1u;                                               // count=1 (valid user D#)
    g0[1] = lds_byte_off;                                     // lds_addr  [63:32]
    g0[2] = (unsigned)ga;                                     // global_addr[95:64]
    g0[3] = (unsigned)((ga >> 32) & 0x01FFFFFFu) | (2u << 30);// addr[56:32] + type=2
    tdm_i32x8 g1;
    g1[0] = (int)(1u << 16);                                  // data_size=1 (2B), mask=0
    g1[1] = (int)((tensor_k & 0xFFFFu) << 16);                // tensor_dim0 lo
    g1[2] = (int)((tensor_k >> 16) | ((tensor_rows & 0xFFFFu) << 16)); // td0 hi | td1 lo
    g1[3] = (int)((tensor_rows >> 16) | (tile_k << 16));      // td1 hi | tile_dim0
    g1[4] = (int)(tile_rows & 0xFFFFu);                       // tile_dim1 (tile_dim2=0)
    g1[5] = (int)(unsigned)(row_stride_elems & 0xFFFFFFFFull);// tensor_dim0_stride lo
    g1[6] = (int)(unsigned)((row_stride_elems >> 32) & 0xFFFFull); // stride hi
    g1[7] = 0;
    tdm_i32x4 z4 = {0, 0, 0, 0};
#if TDM_ARITY6
    tdm_i32x8 z8 = {0, 0, 0, 0, 0, 0, 0, 0};
    __builtin_amdgcn_tensor_load_to_lds(g0, g1, z4, z4, z8, 0);
#else
    __builtin_amdgcn_tensor_load_to_lds(g0, g1, z4, z4, 0);
#endif
}
#endif  // HAVE_TDM

// ---------------------------------------------------------------------------
// f32 -> bf16 conversion (grid-stride)
// ---------------------------------------------------------------------------
__global__ __launch_bounds__(256) void cvt_f32_bf16(const float* __restrict__ s,
                                                    unsigned short* __restrict__ d,
                                                    long long n) {
    long long i = (long long)blockIdx.x * blockDim.x + threadIdx.x;
    long long stride = (long long)gridDim.x * blockDim.x;
    for (; i < n; i += stride) d[i] = f2bf(s[i]);
}

__global__ __launch_bounds__(256) void copy_f32(const float* __restrict__ s,
                                                float* __restrict__ d, int n) {
    int i = blockIdx.x * 256 + threadIdx.x;
    if (i < n) d[i] = s[i];
}

// ---------------------------------------------------------------------------
// Generic bf16 WMMA GEMM:  C = op(A[MxK] * B[KxN] + bias)
//   BCOL  : B stored column-major in global (B[k][n] = Bm[n*ldb + k])
//   RELU  : apply max(x,0)
//   OBF16 : store bf16, else f32
//   CAUSAL: write -inf where col > row; fully-masked blocks skip all MMA work
//
// Block tile 128x128, K-step 32. 8 waves (4 in M x 2 in N), wave tile 32x64
// => 8 v_wmma per wave per k-step. LDS: A row-major, B column-major (so every
// fragment dword is one aligned 32-bit LDS read; merges into ds_load_b128).
// With TDM: A tile (and B tile when BCOL) moved by tensor_load_to_lds,
// double-buffered, one barrier per k-step, fenced by s_wait_tensorcnt.
// ---------------------------------------------------------------------------
#define BM 128
#define BN 128
#define BK 32

template <bool BCOL, bool RELU, bool OBF16, bool CAUSAL>
__global__ __launch_bounds__(256)
void gemm_bf16(const unsigned short* __restrict__ A,
               const unsigned short* __restrict__ Bm,
               const float* __restrict__ bias,
               void* __restrict__ Cp,
               int M, int N, int K, int lda, int ldb, int ldc,
               long long strideA, long long strideB, long long strideC) {
    __shared__ __align__(16) unsigned short As[2][BM][BK];    // 2 x 8 KB, row-major
    __shared__ __align__(16) unsigned short Bsc[2][BN][BK];   // 2 x 8 KB, col-major [n][k]

    const int bz = blockIdx.z;
    A  += (long long)bz * strideA;
    Bm += (long long)bz * strideB;
    float*          Cf = (float*)Cp          + (OBF16 ? 0 : (long long)bz * strideC);
    unsigned short* Cb = (unsigned short*)Cp + (OBF16 ? (long long)bz * strideC : 0);

    const int tid  = threadIdx.x;
    const int wave = tid >> 5;     // 0..7
    const int lane = tid & 31;
    const int half = lane >> 4;    // 0/1
    const int idx  = lane & 15;    // 0..15
    const int wm   = wave >> 1;    // 0..3  (M direction, 32 rows each)
    const int wn   = wave & 1;     // 0..1  (N direction, 64 cols each)
    const int rowBase = blockIdx.y * BM;
    const int colBase = blockIdx.x * BN;

    if (CAUSAL && colBase > rowBase + BM - 1) {
        for (int i = tid; i < BM * BN; i += 256) {
            int r = i / BN, c = i % BN;
            Cf[(long long)(rowBase + r) * ldc + colBase + c] = -__builtin_inff();
        }
        return;
    }

    v8f acc[2][4];
    #pragma unroll
    for (int i = 0; i < 2; ++i)
        #pragma unroll
        for (int j = 0; j < 4; ++j)
            #pragma unroll
            for (int v = 0; v < 8; ++v) acc[i][j][v] = 0.0f;

    // Fragment build + 8 WMMA for one K-step from the given LDS buffers.
    auto compute_tile = [&](const unsigned short (*Asb)[BK],
                            const unsigned short (*Bsb)[BK]) {
        BF16Frag af[2];
        #pragma unroll
        for (int mi = 0; mi < 2; ++mi) {
            int m = wm * 32 + mi * 16 + idx;
            #pragma unroll
            for (int j = 0; j < 8; ++j) {
                int k0 = ((j & 4) ? 16 : 0) + half * 8 + (j & 3) * 2;   // A 16x32 layout
                af[mi].u[j] = *(const unsigned int*)&Asb[m][k0];
            }
        }
        BF16Frag bfr[4];
        #pragma unroll
        for (int ni = 0; ni < 4; ++ni) {
            int n = wn * 64 + ni * 16 + idx;
            #pragma unroll
            for (int j = 0; j < 8; ++j) {
                int k0 = ((j & 4) ? 16 : 0) + half * 8 + (j & 3) * 2;   // B 32x16 layout
                bfr[ni].u[j] = *(const unsigned int*)&Bsb[n][k0];        // single b32 read
            }
        }
        #pragma unroll
        for (int mi = 0; mi < 2; ++mi)
            #pragma unroll
            for (int ni = 0; ni < 4; ++ni)
                acc[mi][ni] = __builtin_amdgcn_wmma_f32_16x16x32_bf16(
                    false, af[mi].v, false, bfr[ni].v, (short)0, acc[mi][ni], false, false);
    };

#if HAVE_TDM
    // ---- TDM double-buffered pipeline: one barrier per k-step ----
    const unsigned ldsA[2] = { (unsigned)(size_t)&As[0][0][0], (unsigned)(size_t)&As[1][0][0] };
    const unsigned ldsB[2] = { (unsigned)(size_t)&Bsc[0][0][0], (unsigned)(size_t)&Bsc[1][0][0] };
    const bool issuer = (wave == 0);
    if (issuer) {
        tdm_load_tile_2d(ldsA[0], A + (long long)rowBase * lda, BK, BM, K, BM, (unsigned)lda);
        if (BCOL)
            tdm_load_tile_2d(ldsB[0], Bm + (long long)colBase * ldb, BK, BN, K, BN, (unsigned)ldb);
    }
    int cur = 0;
    for (int kt = 0; kt < K; kt += BK) {
        if (!BCOL) {
            // row-major global B: manual transpose copy into Bsc[cur]
            int k  = tid >> 3;             // 0..31
            int n0 = (tid & 7) * 16;       // 0,16,...,112
            const unsigned short* src = Bm + (long long)(kt + k) * ldb + colBase + n0;
            unsigned short tmp[16];
            *(uint4*)&tmp[0] = *(const uint4*)(src);
            *(uint4*)&tmp[8] = *(const uint4*)(src + 8);
            #pragma unroll
            for (int e = 0; e < 16; ++e) Bsc[cur][n0 + e][k] = tmp[e];
            if (kt + BK < K) __builtin_prefetch(src + BK * ldb, 0, 0);  // global_prefetch_b8
        }
        if (issuer) __builtin_amdgcn_s_wait_tensorcnt(0);   // tile kt resident
        __syncthreads();                                    // visible to all waves
        if (issuer && kt + BK < K) {                        // overlap next DMA w/ compute
            tdm_load_tile_2d(ldsA[cur ^ 1], A + (long long)rowBase * lda + kt + BK,
                             BK, BM, (unsigned)(K - kt - BK), BM, (unsigned)lda);
            if (BCOL)
                tdm_load_tile_2d(ldsB[cur ^ 1], Bm + (long long)colBase * ldb + kt + BK,
                                 BK, BN, (unsigned)(K - kt - BK), BN, (unsigned)ldb);
        }
        compute_tile(As[cur], Bsc[cur]);
        cur ^= 1;
    }
#else
    // ---- fallback: classic per-thread global->LDS staging (single buffer) ----
    for (int kt = 0; kt < K; kt += BK) {
        {
            int r = tid >> 1;
            int c = (tid & 1) * 16;
            const unsigned short* src = A + (long long)(rowBase + r) * lda + kt + c;
            *(uint4*)&As[0][r][c]     = *(const uint4*)(src);
            *(uint4*)&As[0][r][c + 8] = *(const uint4*)(src + 8);
            if (kt + BK < K) __builtin_prefetch(src + BK, 0, 0);
        }
        if (BCOL) {
            int n  = tid >> 1;
            int k0 = (tid & 1) * 16;
            const unsigned short* src = Bm + (long long)(colBase + n) * ldb + kt + k0;
            *(uint4*)&Bsc[0][n][k0]     = *(const uint4*)(src);
            *(uint4*)&Bsc[0][n][k0 + 8] = *(const uint4*)(src + 8);
        } else {
            int k  = tid >> 3;
            int n0 = (tid & 7) * 16;
            const unsigned short* src = Bm + (long long)(kt + k) * ldb + colBase + n0;
            unsigned short tmp[16];
            *(uint4*)&tmp[0] = *(const uint4*)(src);
            *(uint4*)&tmp[8] = *(const uint4*)(src + 8);
            #pragma unroll
            for (int e = 0; e < 16; ++e) Bsc[0][n0 + e][k] = tmp[e];
        }
        __syncthreads();
        compute_tile(As[0], Bsc[0]);
        __syncthreads();
    }
#endif

    // ---- epilogue: bias + relu / causal mask, store ----
    #pragma unroll
    for (int mi = 0; mi < 2; ++mi) {
        #pragma unroll
        for (int ni = 0; ni < 4; ++ni) {
            int rowT = rowBase + wm * 32 + mi * 16 + half * 8;
            int col  = colBase + wn * 64 + ni * 16 + idx;
            float bv = bias ? bias[col] : 0.0f;
            #pragma unroll
            for (int v = 0; v < 8; ++v) {
                int row = rowT + v;
                float x = acc[mi][ni][v] + bv;
                if (RELU) x = fmaxf(x, 0.0f);
                if (CAUSAL && col > row) x = -__builtin_inff();
                long long o = (long long)row * ldc + col;
                if (OBF16) Cb[o] = f2bf(x);
                else       Cf[o] = x;
            }
        }
    }
}

// ---------------------------------------------------------------------------
// Row softmax over ncols f32 scores -> bf16 probabilities (one block per row)
// ---------------------------------------------------------------------------
__global__ __launch_bounds__(256)
void softmax_rows(const float* __restrict__ S, unsigned short* __restrict__ P, int ncols) {
    long long row = blockIdx.x;
    const float* src = S + row * (long long)ncols;
    unsigned short* dst = P + row * (long long)ncols;
    __shared__ float sred[8];
    __shared__ float bval;
    const int tid = threadIdx.x, wave = tid >> 5, lane = tid & 31;

    float m = -__builtin_inff();
    for (int c = tid; c < ncols; c += 256) m = fmaxf(m, src[c]);
    #pragma unroll
    for (int o = 16; o > 0; o >>= 1) m = fmaxf(m, __shfl_xor(m, o, 32));
    if (lane == 0) sred[wave] = m;
    __syncthreads();
    if (tid == 0) {
        float mm = sred[0];
        #pragma unroll
        for (int i = 1; i < 8; ++i) mm = fmaxf(mm, sred[i]);
        bval = mm;
    }
    __syncthreads();
    const float rowmax = bval;
    __syncthreads();

    float s = 0.0f;
    for (int c = tid; c < ncols; c += 256) s += __expf(src[c] - rowmax);
    #pragma unroll
    for (int o = 16; o > 0; o >>= 1) s += __shfl_xor(s, o, 32);
    if (lane == 0) sred[wave] = s;
    __syncthreads();
    if (tid == 0) {
        float ss = 0.0f;
        #pragma unroll
        for (int i = 0; i < 8; ++i) ss += sred[i];
        bval = 1.0f / ss;
    }
    __syncthreads();
    const float inv = bval;
    for (int c = tid; c < ncols; c += 256) dst[c] = f2bf(__expf(src[c] - rowmax) * inv);
}

// ---------------------------------------------------------------------------
// Head: logits = O[rows x 512] @ W4[512 x 10] + b4, then log_softmax.
// One wave per row; W4 staged in LDS.
// ---------------------------------------------------------------------------
__global__ __launch_bounds__(256)
void head_logsoftmax(const float* __restrict__ O, const float* __restrict__ W4,
                     const float* __restrict__ b4, float* __restrict__ out, int K) {
    __shared__ float w[512 * CDOUT];   // 20 KB
    for (int i = threadIdx.x; i < K * CDOUT; i += 256) w[i] = W4[i];
    __syncthreads();

    const int wave = threadIdx.x >> 5, lane = threadIdx.x & 31;
    long long row = (long long)blockIdx.x * 8 + wave;
    const float* orow = O + row * K;

    float acc[CDOUT];
    #pragma unroll
    for (int n = 0; n < CDOUT; ++n) acc[n] = 0.0f;
    for (int k = lane; k < K; k += 32) {
        float ov = orow[k];
        #pragma unroll
        for (int n = 0; n < CDOUT; ++n) acc[n] = fmaf(ov, w[k * CDOUT + n], acc[n]);
    }
    #pragma unroll
    for (int n = 0; n < CDOUT; ++n)
        #pragma unroll
        for (int o = 16; o > 0; o >>= 1) acc[n] += __shfl_xor(acc[n], o, 32);

    if (lane == 0) {
        float mx = -__builtin_inff();
        #pragma unroll
        for (int n = 0; n < CDOUT; ++n) { acc[n] += b4[n]; mx = fmaxf(mx, acc[n]); }
        float s = 0.0f;
        #pragma unroll
        for (int n = 0; n < CDOUT; ++n) s += __expf(acc[n] - mx);
        float lse = mx + __logf(s);
        #pragma unroll
        for (int n = 0; n < CDOUT; ++n) out[row * CDOUT + n] = acc[n] - lse;
    }
}

// ---------------------------------------------------------------------------
extern "C" void kernel_launch(void* const* d_in, const int* in_sizes, int n_in,
                              void* d_out, int out_size, void* d_ws, size_t ws_size,
                              hipStream_t stream) {
    const float* X    = (const float*)d_in[0];
    const float* game = (const float*)d_in[1];
    const float* user = (const float*)d_in[2];
    const float* W1   = (const float*)d_in[3];
    const float* b1   = (const float*)d_in[4];
    const float* W2   = (const float*)d_in[5];
    const float* b2   = (const float*)d_in[6];
    const float* W3   = (const float*)d_in[7];
    const float* b3   = (const float*)d_in[8];
    const float* W4   = (const float*)d_in[9];
    const float* b4   = (const float*)d_in[10];

    const long long M  = (long long)CB * CS;       // 32768 token rows
    const long long nX  = M * CDIN;                // 16.7M
    const long long nW1 = (long long)CDIN * 2 * CDIN;
    const long long nW2 = (long long)2 * CDIN * CDH;
    const long long nW3 = (long long)CDH * (CDH / 2);
    const long long nH  = M * CDH;
    const long long nS  = (long long)CB * CS * CS;
    const long long nO  = M * (CDH / 2);

    char* ws = (char*)d_ws;
    size_t off = 0;
    auto take = [&](size_t bytes) -> char* {
        char* p = ws + off;
        off = (off + bytes + 255) & ~(size_t)255;
        return p;
    };
    unsigned short* Xb   = (unsigned short*)take(nX * 2);
    unsigned short* W1b  = (unsigned short*)take(nW1 * 2);
    unsigned short* W2b  = (unsigned short*)take(nW2 * 2);
    unsigned short* W3b  = (unsigned short*)take(nW3 * 2);
    unsigned short* H1b  = (unsigned short*)take(nH * 2);   // relu(X W1 + b1)
    unsigned short* Hb   = (unsigned short*)take(nH * 2);   // relu(H1 W2 + b2)
    float*          Sf   = (float*)take(nS * 4);            // raw causal scores
    unsigned short* Pb   = (unsigned short*)take(nS * 2);   // softmax probs
    unsigned short* CTXb = (unsigned short*)take(nH * 2);   // attn output
    float*          Of   = (float*)take(nO * 4);            // relu(ctx W3 + b3)

    // 1) convert to bf16
    cvt_f32_bf16<<<2048, 256, 0, stream>>>(X,  Xb,  nX);
    cvt_f32_bf16<<<512,  256, 0, stream>>>(W1, W1b, nW1);
    cvt_f32_bf16<<<512,  256, 0, stream>>>(W2, W2b, nW2);
    cvt_f32_bf16<<<512,  256, 0, stream>>>(W3, W3b, nW3);

    // 2) H1 = relu(X @ W1 + b1)   [32768 x 1024], K=512
    gemm_bf16<false, true, true, false><<<dim3(2 * CDIN / BN, M / BM, 1), 256, 0, stream>>>(
        Xb, W1b, b1, H1b, (int)M, 2 * CDIN, CDIN, CDIN, 2 * CDIN, 2 * CDIN, 0, 0, 0);

    // 3) H = relu(H1 @ W2 + b2)   [32768 x 1024], K=1024
    gemm_bf16<false, true, true, false><<<dim3(CDH / BN, M / BM, 1), 256, 0, stream>>>(
        H1b, W2b, b2, Hb, (int)M, CDH, 2 * CDIN, 2 * CDIN, CDH, CDH, 0, 0, 0);

    // 4) S = H H^T with causal mask (per batch): [2048 x 2048], K=1024
    gemm_bf16<true, false, false, true><<<dim3(CS / BN, CS / BM, CB), 256, 0, stream>>>(
        Hb, Hb, nullptr, Sf, CS, CS, CDH, CDH, CDH, CS,
        (long long)CS * CDH, (long long)CS * CDH, (long long)CS * CS);

    // 5) P = softmax_rows(S)  -> bf16
    softmax_rows<<<CB * CS, 256, 0, stream>>>(Sf, Pb, CS);

    // 6) CTX = P @ H (per batch): [2048 x 1024], K=2048
    gemm_bf16<false, false, true, false><<<dim3(CDH / BN, CS / BM, CB), 256, 0, stream>>>(
        Pb, Hb, nullptr, CTXb, CS, CDH, CS, CS, CDH, CDH,
        (long long)CS * CS, (long long)CS * CDH, (long long)CS * CDH);

    // 7) O = relu(CTX @ W3 + b3)  [32768 x 512], K=1024  (f32 out)
    gemm_bf16<false, true, false, false><<<dim3((CDH / 2) / BN, M / BM, 1), 256, 0, stream>>>(
        CTXb, W3b, b3, Of, (int)M, CDH / 2, CDH, CDH, CDH / 2, CDH / 2, 0, 0, 0);

    // 8) out = log_softmax(O @ W4 + b4)
    float* outF = (float*)d_out;
    head_logsoftmax<<<M / 8, 256, 0, stream>>>(Of, W4, b4, outF, CDH / 2);

    // 9) pass-through tuple members: game_vector, user_vector
    const int nGV = CB * 2 * CDH;  // 32768 each
    copy_f32<<<(nGV + 255) / 256, 256, 0, stream>>>(game, outF + (long long)M * CDOUT, nGV);
    copy_f32<<<(nGV + 255) / 256, 256, 0, stream>>>(user, outF + (long long)M * CDOUT + nGV, nGV);
}
